// W8A8BFP32OFP32LinearForMac_61297773248989
// MI455X (gfx1250) — 8-bit matrix operand order repaired and checked by reading, compiled, not run on hardware
//
#include <hip/hip_runtime.h>

typedef __attribute__((ext_vector_type(8))) int v8i;
typedef __attribute__((ext_vector_type(4))) int i4v;

#define T_M 8192
#define K_F 4096
#define N_F 4096

#define BM 128
#define BN 128
#define BK 64
#define KSTEPS (K_F / BK)

#define LDSTR 20

__device__ __forceinline__ unsigned int pack4(int a, int b, int c, int d) {
  unsigned int ab = __builtin_amdgcn_perm((unsigned)b, (unsigned)a, 0x00000400u);
  unsigned int cd = __builtin_amdgcn_perm((unsigned)d, (unsigned)c, 0x00000400u);
  return __builtin_amdgcn_perm(cd, ab, 0x05040100u);
}

__global__ __launch_bounds__(256) void w8a8_i8gemm(
    const int* __restrict__ x, const int* __restrict__ wgt,
    const int* __restrict__ bias, const float* __restrict__ scales,
    float* __restrict__ out)
{
  __shared__ __align__(16) unsigned int ldsA[2][BM * LDSTR];
  __shared__ __align__(16) unsigned int ldsB[2][BN * LDSTR];

  const int tid  = threadIdx.x;
  const int lane = tid & 31;
  const int wave = tid >> 5;
  const int hi   = lane >> 4;
  const int ln   = lane & 15;
  const int wm   = wave & 3;
  const int wn   = wave >> 2;

  const int bn = blockIdx.x * BN;
  const int bm = blockIdx.y * BM;

  int aRow[8], aKd[8], aOff[8];
  #pragma unroll
  for (int i = 0; i < 8; ++i) {
    int f = tid + 256 * i;
    aRow[i] = f >> 4; aKd[i] = f & 15;
    aOff[i] = aRow[i] * K_F + aKd[i] * 4;
  }
  int bNg[2], bKg[2], bOff[2][4];
  #pragma unroll
  for (int i = 0; i < 2; ++i) {
    int f = tid + 256 * i;
    bNg[i] = f & 31; bKg[i] = f >> 5;
    #pragma unroll
    for (int j = 0; j < 4; ++j)
      bOff[i][j] = (bKg[i] * 4 + j) * N_F + bNg[i] * 4;
  }

  const int* xA = x + (long)bm * K_F;
  const int* wB = wgt + bn;

  i4v gA[8];
  i4v gB[2][4];

  auto loadStage = [&](int k0) {
    const int* pa = xA + k0;
    const int* pb = wB + (long)k0 * N_F;
    #pragma unroll
    for (int i = 0; i < 8; ++i)
      gA[i] = *(const i4v*)(pa + aOff[i]);
    #pragma unroll
    for (int i = 0; i < 2; ++i) {
      #pragma unroll
      for (int j = 0; j < 4; ++j)
        gB[i][j] = *(const i4v*)(pb + bOff[i][j]);
    }
  };

  v8i acc[2][4];
  #pragma unroll
  for (int am = 0; am < 2; ++am)
    #pragma unroll
    for (int s = 0; s < 4; ++s)
      #pragma unroll
      for (int r = 0; r < 8; ++r) acc[am][s][r] = 0;

  loadStage(0);

  int buf = 0;
  for (int kt = 0; kt < KSTEPS; ++kt) {
    #pragma unroll
    for (int i = 0; i < 8; ++i)
      ldsA[buf][aRow[i] * LDSTR + aKd[i]] = pack4(gA[i].x, gA[i].y, gA[i].z, gA[i].w);
    #pragma unroll
    for (int i = 0; i < 2; ++i) {
      #pragma unroll
      for (int nn = 0; nn < 4; ++nn)
        ldsB[buf][(bNg[i] * 4 + nn) * LDSTR + bKg[i]] =
            pack4(gB[i][0][nn], gB[i][1][nn], gB[i][2][nn], gB[i][3][nn]);
    }
    __syncthreads();

    if (kt + 1 < KSTEPS) loadStage((kt + 1) * BK);

    v8i afrag[2];
    #pragma unroll
    for (int am = 0; am < 2; ++am) {
      const unsigned int* pa = &ldsA[buf][(wm * 32 + am * 16 + ln) * LDSTR + hi * 4];
      uint2 t0 = *(const uint2*)(pa + 0);
      uint2 t1 = *(const uint2*)(pa + 2);
      uint2 t2 = *(const uint2*)(pa + 8);
      uint2 t3 = *(const uint2*)(pa + 10);
      afrag[am][0] = (int)t0.x; afrag[am][1] = (int)t0.y;
      afrag[am][2] = (int)t1.x; afrag[am][3] = (int)t1.y;
      afrag[am][4] = (int)t2.x; afrag[am][5] = (int)t2.y;
      afrag[am][6] = (int)t3.x; afrag[am][7] = (int)t3.y;
    }

    #pragma unroll
    for (int s = 0; s < 4; ++s) {
      const unsigned int* pb = &ldsB[buf][(wn * 64 + s * 16 + ln) * LDSTR + hi * 4];
      uint4 b0 = *(const uint4*)(pb + 0);
      uint4 b1 = *(const uint4*)(pb + 8);
      v8i bfrag;
      bfrag[0] = (int)b0.x; bfrag[1] = (int)b0.y; bfrag[2] = (int)b0.z; bfrag[3] = (int)b0.w;
      bfrag[4] = (int)b1.x; bfrag[5] = (int)b1.y; bfrag[6] = (int)b1.z; bfrag[7] = (int)b1.w;

      #pragma unroll
      for (int am = 0; am < 2; ++am)
        acc[am][s] = __builtin_amdgcn_wmma_i32_16x16x64_iu8(
            false, afrag[am], true, bfrag, acc[am][s], false, false);
    }

    buf ^= 1;
  }

  #pragma unroll
  for (int s = 0; s < 4; ++s) {
    const int col  = bn + wn * 64 + s * 16 + ln;
    const float sc = scales[col];
    const int   bs = bias[col];
    #pragma unroll
    for (int am = 0; am < 2; ++am) {
      #pragma unroll
      for (int r = 0; r < 8; ++r) {
        const int m = bm + wm * 32 + am * 16 + r + hi * 8;
        out[(long)m * N_F + col] = (float)(acc[am][s][r] + bs) * sc;
      }
    }
  }
}

extern "C" void kernel_launch(void* const* d_in, const int* in_sizes, int n_in,
                              void* d_out, int out_size, void* d_ws, size_t ws_size,
                              hipStream_t stream) {
  (void)in_sizes; (void)n_in; (void)out_size; (void)d_ws; (void)ws_size;
  const int*   x      = (const int*)d_in[0];
  const int*   wgt    = (const int*)d_in[1];
  const int*   bias   = (const int*)d_in[2];
  const float* scales = (const float*)d_in[3];
  float*       out    = (float*)d_out;

  dim3 grid(N_F / BN, T_M / BM);
  w8a8_i8gemm<<<grid, dim3(256), 0, stream>>>(x, wgt, bias, scales, out);
}
